// EuclideanCodebook_1133871366375
// MI455X (gfx1250) — compile-verified, matrix operand order checked by reading, not run on hardware
//
#include <hip/hip_runtime.h>
#include <hip/hip_bf16.h>
#include <math.h>

// ---------------------------------------------------------------------------
// VQ EuclideanCodebook for MI455X (gfx1250, wave32, WMMA).
//  - sim GEMM (65536x2048x256) via v_wmma_f32_16x16x32_bf16, f32 accumulate
//  - codebook tiles double-buffered in LDS via global_load_async_to_lds_b128
//    (ASYNCcnt), shared by all 4 waves of the block; B read back via ds_load
//  - argmax fused into the GEMM epilogue (per-lane running max + shfl reduce)
//  - gather + scatter-add fused in the same wave
// ---------------------------------------------------------------------------

#define DIMX 256
#define CB   2048
#define NT   (CB / 16)        // 128 column tiles
#define DECAY 0.99f
#define EPSILON 1e-5f
#define TILE_BYTES 8192       // 16 codes * 256 bf16 * 2B

typedef __attribute__((ext_vector_type(16))) __bf16 v16bf;
typedef __attribute__((ext_vector_type(8)))  __bf16 v8bf;
typedef __attribute__((ext_vector_type(8)))  float  v8f;
typedef __attribute__((ext_vector_type(4)))  float  v4f;

// ---------------- K0: zero scratch (counts, total, embed_sum) --------------
__global__ void k_zero(float* __restrict__ p, int n) {
    int i = blockIdx.x * blockDim.x + threadIdx.x;
    if (i < n) p[i] = 0.0f;
}

// ---------------- K1: L2-normalize codebook rows -> bf16 -------------------
__global__ void k_norm_codebook(const float* __restrict__ embed,
                                __bf16* __restrict__ cn) {
    const int lane = threadIdx.x & 31;
    const int row  = blockIdx.x * (blockDim.x >> 5) + (threadIdx.x >> 5);
    const float* e = embed + (size_t)row * DIMX;
    float v[8];
    float ss = 0.0f;
#pragma unroll
    for (int i = 0; i < 8; ++i) {
        v[i] = e[lane + i * 32];
        ss += v[i] * v[i];
    }
#pragma unroll
    for (int off = 16; off >= 1; off >>= 1)
        ss += __shfl_xor(ss, off, 32);
    const float scale = 1.0f / fmaxf(sqrtf(ss), 1e-12f);
    __bf16* o = cn + (size_t)row * DIMX;
#pragma unroll
    for (int i = 0; i < 8; ++i)
        o[lane + i * 32] = (__bf16)(v[i] * scale);
}

// ---------------- K2: fused GEMM + argmax + gather + scatter ---------------
__global__ void k_sim_argmax(const float*  __restrict__ x,
                             const __bf16* __restrict__ cn,
                             const float*  __restrict__ embed,
                             float* __restrict__ quantized,
                             float* __restrict__ embed_ind,
                             float* __restrict__ counts,
                             float* __restrict__ embed_sum) {
    // Single shared object in this kernel -> LDS offset 0.
    // 2 x 8KB double buffer + 16B sentinel (never touched by the async DMA).
    __shared__ alignas(16) char ldsRaw[2 * TILE_BYTES + 16];

    const int tid     = threadIdx.x;           // 0..127 (4 waves)
    const int lane    = tid & 31;
    const int wave    = tid >> 5;
    const int rowBase = (blockIdx.x * (blockDim.x >> 5) + wave) * 16;
    const int half    = lane >> 4;   // which K sub-stripe this lane holds
    const int m       = lane & 15;   // row (A) / column (B) within tile

    // Visible store into the LDS global (sentinel region) so GlobalOpt does
    // not treat it as never-stored and fold the ds_loads to undef — the real
    // writes come from inline-asm async DMA which the optimizer cannot see.
    if (tid == 0) ldsRaw[2 * TILE_BYTES] = 0;

    // ---- async stage of one 16-code tile into LDS (all 128 threads) ----
    // Each thread copies 4 x 16B; per-wave ASYNCcnt += 4 per stage.
    auto stageTile = [&](int ct, int buf) {
        const char* gsrc = (const char*)cn + (size_t)ct * TILE_BYTES;
#pragma unroll
        for (int j = 0; j < 4; ++j) {
            const unsigned byteOff = (unsigned)(tid * 16 + j * 2048);
            const unsigned ldsOff  = (unsigned)(buf * TILE_BYTES) + byteOff;
            const char* g = gsrc + byteOff;
            asm volatile("global_load_async_to_lds_b128 %[lds], %[gaddr], off"
                         :: [lds] "v"(ldsOff), [gaddr] "v"(g)
                         : "memory");
        }
    };

    // ---- load A: row (rowBase+m), K stripes per ISA 16-bit A layout ----
    v16bf a[8];
    {
        const float* xr = x + (size_t)(rowBase + m) * DIMX;
#pragma unroll
        for (int kc = 0; kc < 8; ++kc) {
            const int k0 = kc * 32 + half * 8;
            v16bf av;
#pragma unroll
            for (int i = 0; i < 8; ++i) {
                av[i]     = (__bf16)xr[k0 + i];
                av[8 + i] = (__bf16)xr[k0 + 16 + i];
            }
            a[kc] = av;
        }
    }

    float bestV[8];
    int   bestI[8];
#pragma unroll
    for (int v = 0; v < 8; ++v) { bestV[v] = -3.4e38f; bestI[v] = 0; }

    stageTile(0, 0);

    for (int ct = 0; ct < NT; ++ct) {
        const int buf = ct & 1;
        if (ct + 1 < NT) {
            stageTile(ct + 1, buf ^ 1);
            // oldest 4 asyncs (tile ct) done when <=4 remain outstanding
            asm volatile("s_wait_asynccnt 4" ::: "memory");
        } else {
            asm volatile("s_wait_asynccnt 0" ::: "memory");
        }
        __syncthreads();   // all waves' quarter-copies of tile ct visible

        // gather all 8 B chunks from LDS (ds_load_b128 pairs), then 8 WMMAs
        const __bf16* bb =
            (const __bf16*)(ldsRaw + buf * TILE_BYTES + m * (DIMX * 2));
        v16bf bv[8];
#pragma unroll
        for (int kc = 0; kc < 8; ++kc) {
            const int k0 = kc * 32 + half * 8;
            v8bf lo = *(const v8bf*)(bb + k0);
            v8bf hi = *(const v8bf*)(bb + k0 + 16);
            v16bf b;
#pragma unroll
            for (int i = 0; i < 8; ++i) { b[i] = lo[i]; b[8 + i] = hi[i]; }
            bv[kc] = b;
        }
        v8f acc = {0.f, 0.f, 0.f, 0.f, 0.f, 0.f, 0.f, 0.f};
#pragma unroll
        for (int kc = 0; kc < 8; ++kc) {
            acc = __builtin_amdgcn_wmma_f32_16x16x32_bf16(
                false, a[kc], false, bv[kc], (short)0, acc, false, false);
        }

        const int col = ct * 16 + m;
#pragma unroll
        for (int v = 0; v < 8; ++v) {
            const float f = acc[v];
            if (f > bestV[v]) { bestV[v] = f; bestI[v] = col; }
        }
        __syncthreads();   // done reading buf before it is recycled
    }

    // ---- reduce (max, argmax) across the 16 lanes of each wave half ----
#pragma unroll
    for (int off = 8; off >= 1; off >>= 1) {
#pragma unroll
        for (int v = 0; v < 8; ++v) {
            const float ov = __shfl_xor(bestV[v], off, 32);
            const int   oi = __shfl_xor(bestI[v], off, 32);
            if (ov > bestV[v] || (ov == bestV[v] && oi < bestI[v])) {
                bestV[v] = ov; bestI[v] = oi;
            }
        }
    }

    // ---- per-row epilogue: emit idx, histogram, gather, scatter-add ----
    for (int r = 0; r < 16; ++r) {
        const int idx = __shfl(bestI[r & 7], (r >> 3) * 16, 32);
        const int row = rowBase + r;
        if (lane == 0) {
            embed_ind[row] = (float)idx;
            atomicAdd(&counts[idx], 1.0f);
        }
        // quantized[row] = embed[idx]  (64 float4 per row, 2 per lane)
        const v4f* src = (const v4f*)(embed + (size_t)idx * DIMX);
        v4f*       dst = (v4f*)(quantized + (size_t)row * DIMX);
        dst[lane]      = src[lane];
        dst[lane + 32] = src[lane + 32];
        // embed_sum[idx] += x[row]  (8 scalar f32 atomics per lane)
        const float* xr = x + (size_t)row * DIMX;
        float*       es = embed_sum + (size_t)idx * DIMX;
#pragma unroll
        for (int e = 0; e < 8; ++e) {
            const int c = lane + e * 32;
            atomicAdd(&es[c], xr[c]);
        }
    }
}

// ---------------- K3: EMA cluster size + total reduction -------------------
__global__ void k_ema_cs(const float* __restrict__ cluster_size,
                         const float* __restrict__ counts,
                         float* __restrict__ new_cs,
                         float* __restrict__ total) {
    const int c    = blockIdx.x * blockDim.x + threadIdx.x;
    const int lane = threadIdx.x & 31;
    const int wave = threadIdx.x >> 5;
    float ncs = cluster_size[c] * DECAY + counts[c] * (1.0f - DECAY);
    new_cs[c] = ncs;
    float s = ncs;
#pragma unroll
    for (int off = 16; off >= 1; off >>= 1) s += __shfl_xor(s, off, 32);
    __shared__ float red[8];
    if (lane == 0) red[wave] = s;
    __syncthreads();
    if (wave == 0) {
        float t = (lane < (int)(blockDim.x >> 5)) ? red[lane] : 0.0f;
#pragma unroll
        for (int off = 4; off >= 1; off >>= 1) t += __shfl_xor(t, off, 32);
        if (lane == 0) atomicAdd(total, t);
    }
}

// ---------------- K4: EMA embed_avg + laplace-smoothed renormalize ---------
__global__ void k_ema_embed(const float* __restrict__ embed_avg,
                            const float* __restrict__ embed_sum,
                            const float* __restrict__ new_cs,
                            const float* __restrict__ total,
                            float* __restrict__ new_embed_avg,
                            float* __restrict__ new_embed) {
    const int lane = threadIdx.x & 31;
    const int row  = blockIdx.x * (blockDim.x >> 5) + (threadIdx.x >> 5);
    const float tot = *total;
    const float cs  = (new_cs[row] + EPSILON) /
                      (tot + (float)CB * EPSILON) * tot;
    const float inv_cs = 1.0f / cs;
    const size_t base = (size_t)row * DIMX;
    float w[8];
    float ss = 0.0f;
#pragma unroll
    for (int i = 0; i < 8; ++i) {
        const int c = lane + i * 32;
        const float nea = embed_avg[base + c] * DECAY +
                          embed_sum[base + c] * (1.0f - DECAY);
        new_embed_avg[base + c] = nea;
        w[i] = nea * inv_cs;
        ss += w[i] * w[i];
    }
#pragma unroll
    for (int off = 16; off >= 1; off >>= 1) ss += __shfl_xor(ss, off, 32);
    const float scale = 1.0f / fmaxf(sqrtf(ss), 1e-12f);
#pragma unroll
    for (int i = 0; i < 8; ++i)
        new_embed[base + lane + i * 32] = w[i] * scale;
}

// ---------------------------------------------------------------------------
extern "C" void kernel_launch(void* const* d_in, const int* in_sizes, int n_in,
                              void* d_out, int out_size, void* d_ws, size_t ws_size,
                              hipStream_t stream) {
    const float* x            = (const float*)d_in[0];  // [16,4096,256]
    const float* embed        = (const float*)d_in[1];  // [2048,256]
    const float* cluster_size = (const float*)d_in[2];  // [2048]
    const float* embed_avg    = (const float*)d_in[3];  // [2048,256]

    const int N = 16 * 4096;  // 65536 rows

    // d_out: quantized | embed_ind | new_embed | new_cluster_size | new_embed_avg
    float* out_quant = (float*)d_out;
    float* out_ind   = out_quant + (size_t)N * DIMX;          // 16777216
    float* out_embed = out_ind + N;                           // 16842752
    float* out_cs    = out_embed + (size_t)CB * DIMX;         // 17367040
    float* out_avg   = out_cs + CB;                           // 17369088

    // workspace: cn_bf16 (1MB) | counts[2048] | total[1] | embed_sum[CB*DIMX]
    float*  wsf       = (float*)d_ws;
    __bf16* cn        = (__bf16*)d_ws;          // CB*DIMX bf16 = 262144 floats
    float*  counts    = wsf + 262144;
    float*  total     = wsf + 264192;
    float*  embed_sum = wsf + 264256;

    // K0: zero the accumulation region [counts .. embed_sum end)
    {
        const int nz = (264256 + CB * DIMX) - 262144;
        k_zero<<<(nz + 255) / 256, 256, 0, stream>>>(wsf + 262144, nz);
    }

    // K1: normalized bf16 codebook (one wave per row, 8 waves/block)
    k_norm_codebook<<<CB / 8, 256, 0, stream>>>(embed, cn);

    // K2: WMMA similarity GEMM + argmax + gather + scatter
    // 16 rows per wave, 4 waves per block -> 64 rows/block, 1024 blocks
    k_sim_argmax<<<N / 64, 128, 0, stream>>>(x, cn, embed, out_quant, out_ind,
                                             counts, embed_sum);

    // K3: EMA cluster sizes + grand total
    k_ema_cs<<<CB / 256, 256, 0, stream>>>(cluster_size, counts, out_cs, total);

    // K4: EMA embed_avg + laplace-smoothed renormalized codebook
    k_ema_embed<<<CB / 8, 256, 0, stream>>>(embed_avg, embed_sum, out_cs, total,
                                            out_avg, out_embed);
}
